// LmsPAM_Module_38233798869169
// MI455X (gfx1250) — matrix operand order repaired and checked by reading, not compiled, not run on hardware
//
#include <hip/hip_runtime.h>

// ---------------------------------------------------------------------------
// LmsPAM linear-attention block for MI455X (gfx1250), bf16 WMMA pipeline.
// B=8, C=512, M=64, N=H*W=4096.
// ---------------------------------------------------------------------------

typedef __attribute__((ext_vector_type(16))) __bf16 v16bf;
typedef __attribute__((ext_vector_type(8)))  __bf16 v8bf;
typedef __attribute__((ext_vector_type(4)))  __bf16 v4bf;
typedef __attribute__((ext_vector_type(8)))  float  v8f;
typedef __attribute__((ext_vector_type(4)))  unsigned u32x4;
typedef __attribute__((ext_vector_type(8)))  int      i32x8;
typedef __attribute__((ext_vector_type(4)))  int      i32x4;

#define NPIX   4096
#define CCH    512
#define MCH    64
#define NBATCH 8

static constexpr int LDK = 40;   // padded k-stride for LDS tiles ([row][k] layout)
static constexpr float EPS_C = 1e-10f;

// Native converts -> v_cvt_*bf16_f32 on gfx1250
__device__ __forceinline__ __bf16 f2bf(float f) { return (__bf16)f; }
__device__ __forceinline__ float bf2f(__bf16 v) { return (float)v; }

// elus feature map: 10*relu(t) + exp(10*min(t,0))
__device__ __forceinline__ float elus(float t) {
  return 10.0f * fmaxf(t, 0.0f) + __expf(10.0f * fminf(t, 0.0f));
}

// ---------------------------------------------------------------------------
// TDM: async 2D tile DMA global->LDS (tile_k bf16 per row, tile_rows rows,
// row_stride elements between rows). pad_enable inserts 4 DWORDs (8 bf16)
// after every 16 DWORDs (32 bf16) -> reproduces the LDK=40 padded layout.
// Tracked by TENSORcnt (in-order per wave).
// ---------------------------------------------------------------------------
__device__ __forceinline__ void tdm_load_tile_2d(unsigned lds_addr, const __bf16* gptr,
                                                 unsigned tile_rows, unsigned tile_k,
                                                 unsigned row_stride) {
  unsigned long long ga = (unsigned long long)(uintptr_t)gptr;
  u32x4 g0;
  g0[0] = 1u;                                               // count=1, user mode
  g0[1] = lds_addr;                                         // lds_addr [63:32]
  g0[2] = (unsigned)ga;                                     // global_addr lo
  g0[3] = (unsigned)((ga >> 32) & 0x1FFFFFFu) | (2u << 30); // global_addr hi | type=2
  i32x8 g1;
  g1[0] = (int)((1u << 16)      // data_size = 2 bytes
              | (1u << 20)      // pad_enable
              | (3u << 22)      // pad_interval: 16 DWORDs
              | (3u << 25));    // pad_amount: 4 DWORDs
  g1[1] = (int)((row_stride & 0xFFFFu) << 16);                       // tensor_dim0 lo16
  g1[2] = (int)((row_stride >> 16) | ((tile_rows & 0xFFFFu) << 16)); // dim0 hi | tensor_dim1 lo
  g1[3] = (int)((tile_rows >> 16) | (tile_k << 16));                 // dim1 hi | tile_dim0
  g1[4] = (int)(tile_rows & 0xFFFFu);                                // tile_dim1 (tile_dim2=0)
  g1[5] = (int)row_stride;                                           // tensor_dim0_stride lo32
  g1[6] = 0;
  g1[7] = 0;
  i32x4 z4 = (i32x4)0;
#if __clang_major__ >= 23
  i32x8 z8 = (i32x8)0;
  __builtin_amdgcn_tensor_load_to_lds(g0, g1, z4, z4, z8, 0);
#else
  __builtin_amdgcn_tensor_load_to_lds(g0, g1, z4, z4, 0);
#endif
}

// One BK=32 WMMA step. As: [BM][LDK] (m-major), Bs: [BN][LDK] (n-major), bf16.
// Fragments per ISA 7.12.2; LDK=40 keeps every chunk 16B-aligned -> ds_load_b128.
template <int MT, int NT>
__device__ __forceinline__ void mma_step(const __bf16* As, const __bf16* Bs,
                                         int m0, int n0, v8f (&acc)[MT][NT]) {
  const int lane = threadIdx.x & 31;
  const int g    = lane >> 4;
  const int lr   = lane & 15;
  v16bf a[MT], b[NT];
#pragma unroll
  for (int i = 0; i < MT; ++i) {
    const __bf16* ap = As + (m0 + i * 16 + lr) * LDK;
    v8bf lo = *(const v8bf*)(ap + g * 8);
    v8bf hi = *(const v8bf*)(ap + 16 + g * 8);
    a[i] = __builtin_shufflevector(lo, hi, 0,1,2,3,4,5,6,7,8,9,10,11,12,13,14,15);
  }
#pragma unroll
  for (int j = 0; j < NT; ++j) {
    const __bf16* bp = Bs + (n0 + j * 16 + lr) * LDK + g * 8;
    v8bf lo = *(const v8bf*)(bp);
    v8bf hi = *(const v8bf*)(bp + 16);
    b[j] = __builtin_shufflevector(lo, hi, 0,1,2,3,4,5,6,7,8,9,10,11,12,13,14,15);
  }
#pragma unroll
  for (int i = 0; i < MT; ++i)
#pragma unroll
    for (int j = 0; j < NT; ++j)
      acc[i][j] = __builtin_amdgcn_wmma_f32_16x16x32_bf16(
          false, a[i], false, b[j], (short)0, acc[i][j], false, false);
}

// ---------------------------------------------------------------------------
// Projection GEMM: Out_bf16[b] = act(W (Cout x Cin) @ X[b] (Cin x N) + bias)
// ---------------------------------------------------------------------------
template <int BM, int BN, bool ELUS>
__global__ __launch_bounds__(256) void gemm_wx_kernel(
    const float* __restrict__ W, const float* __restrict__ X,
    const float* __restrict__ bias, __bf16* __restrict__ Out,
    int Cout, int Cin) {
  constexpr int BK      = 32;
  constexpr int WAVES_M = BM / 32;
  constexpr int WAVES_N = 8 / WAVES_M;
  constexpr int MT      = 2;
  constexpr int NT      = (BN / WAVES_N) / 16;
  constexpr int BN4     = BN / 4;
  __shared__ alignas(16) __bf16 As[BM * LDK];
  __shared__ alignas(16) __bf16 Bs[BN * LDK];

  const int bz   = blockIdx.z;
  const int row0 = blockIdx.y * BM;
  const int col0 = blockIdx.x * BN;
  const int tid  = threadIdx.x;
  const int wave = tid >> 5;
  const int m0   = (wave / WAVES_N) * 32;
  const int n0   = (wave % WAVES_N) * (16 * NT);
  const float* Xb = X + (size_t)bz * Cin * NPIX;

  v8f acc[MT][NT];
#pragma unroll
  for (int i = 0; i < MT; ++i)
#pragma unroll
    for (int j = 0; j < NT; ++j)
#pragma unroll
      for (int r = 0; r < 8; ++r) acc[i][j][r] = 0.0f;

  for (int kk = 0; kk < Cin; kk += BK) {
    // A tile: float4 global loads -> v4bf LDS stores (k-contiguous)
    for (int idx = tid; idx < BM * (BK / 4); idx += 256) {
      int r = idx >> 3, k4 = (idx & 7) << 2;
      float4 w4 = *(const float4*)(W + (size_t)(row0 + r) * Cin + kk + k4);
      v4bf o; o[0] = f2bf(w4.x); o[1] = f2bf(w4.y); o[2] = f2bf(w4.z); o[3] = f2bf(w4.w);
      *(v4bf*)(As + r * LDK + k4) = o;
    }
    // B tile: float4 global loads over n, scalar LDS stores ([n][k] layout)
    for (int idx = tid; idx < BK * BN4; idx += 256) {
      int k = idx / BN4, n4 = (idx % BN4) << 2;
      float4 x4 = *(const float4*)(Xb + (size_t)(kk + k) * NPIX + col0 + n4);
      Bs[(n4 + 0) * LDK + k] = f2bf(x4.x);
      Bs[(n4 + 1) * LDK + k] = f2bf(x4.y);
      Bs[(n4 + 2) * LDK + k] = f2bf(x4.z);
      Bs[(n4 + 3) * LDK + k] = f2bf(x4.w);
    }
    if (kk + BK < Cin) {  // prefetch next k-tile -> global_prefetch_b8
      __builtin_prefetch(W + (size_t)(row0 + (tid >> 3)) * Cin + kk + BK + ((tid & 7) << 2), 0, 1);
      __builtin_prefetch(Xb + (size_t)(kk + BK + (tid >> 6)) * NPIX + col0 + ((tid & 63) << 1), 0, 1);
    }
    __syncthreads();
    mma_step<MT, NT>(As, Bs, m0, n0, acc);
    __syncthreads();
  }

  const int lane = tid & 31, g = lane >> 4, lr = lane & 15;
  __bf16* Ob = Out + (size_t)bz * Cout * NPIX;
#pragma unroll
  for (int i = 0; i < MT; ++i)
#pragma unroll
    for (int j = 0; j < NT; ++j)
#pragma unroll
      for (int r = 0; r < 8; ++r) {
        int row = row0 + m0 + i * 16 + g * 8 + r;
        int col = col0 + n0 + j * 16 + lr;
        float v = acc[i][j][r] + bias[row];
        if (ELUS) v = elus(v);
        Ob[(size_t)row * NPIX + col] = f2bf(v);
      }
}

// ---------------------------------------------------------------------------
// Ksum[row] = sum_n K[row][n] + EPS   (row in [0, 3*B*64))
// ---------------------------------------------------------------------------
__global__ __launch_bounds__(128) void ksum_kernel(const __bf16* __restrict__ K,
                                                   float* __restrict__ Ksum) {
  __shared__ float red[128];
  const int row = blockIdx.x;
  float s = 0.0f;
  for (int n = threadIdx.x * 8; n < NPIX; n += 128 * 8) {
    v8bf kv = *(const v8bf*)(K + (size_t)row * NPIX + n);
#pragma unroll
    for (int i = 0; i < 8; ++i) s += bf2f(kv[i]);
  }
  red[threadIdx.x] = s;
  __syncthreads();
  for (int o = 64; o > 0; o >>= 1) {
    if (threadIdx.x < o) red[threadIdx.x] += red[threadIdx.x + o];
    __syncthreads();
  }
  if (threadIdx.x == 0) Ksum[row] = red[0] + EPS_C;
}

// ---------------------------------------------------------------------------
// normv[br,b,n] = 1 / sum_m Q[b,m,n] * Ksum[br,b,m]
// ---------------------------------------------------------------------------
__global__ __launch_bounds__(256) void norm_kernel(const __bf16* __restrict__ Q,
                                                   const float* __restrict__ Ks,
                                                   float* __restrict__ normv) {
  const int gid = blockIdx.x * 256 + threadIdx.x;  // = (br*8 + b)*4096 + n
  const int n   = gid & (NPIX - 1);
  const int bb  = gid >> 12;        // br*8 + b
  const int b   = bb & 7;
  float s = 0.0f;
#pragma unroll 8
  for (int m = 0; m < MCH; ++m)
    s += bf2f(Q[((size_t)b * MCH + m) * NPIX + n]) * Ks[bb * MCH + m];
  normv[gid] = 1.0f / s;
}

// ---------------------------------------------------------------------------
// KVt[br,b] (C x M) = V[b] (C x N) @ K_br[b]^T (N x M)
// Tiles staged by the Tensor Data Mover (async DMA, TENSORcnt), double-buffered.
// ---------------------------------------------------------------------------
__global__ __launch_bounds__(256) void gemm_kvt_kernel(
    const __bf16* __restrict__ V, const __bf16* __restrict__ K3,
    __bf16* __restrict__ KVt) {
  constexpr int BM = 128, BN = 64, BK = 32, MT = 2, NT = 2, WAVES_N = 2;
  __shared__ alignas(16) __bf16 As[2][BM * LDK];
  __shared__ alignas(16) __bf16 Bs[2][BN * LDK];

  const int row0 = blockIdx.x * BM;
  const int br   = blockIdx.y;
  const int bz   = blockIdx.z;
  const int tid  = threadIdx.x;
  const int wave = tid >> 5;
  const int m0   = (wave / WAVES_N) * 32;
  const int n0   = (wave % WAVES_N) * 32;
  const __bf16* Vb = V + (size_t)bz * CCH * NPIX + (size_t)row0 * NPIX;
  const __bf16* Kb = K3 + ((size_t)br * NBATCH + bz) * MCH * NPIX;

  const unsigned ldsA[2] = {(unsigned)(uintptr_t)&As[0][0], (unsigned)(uintptr_t)&As[1][0]};
  const unsigned ldsB[2] = {(unsigned)(uintptr_t)&Bs[0][0], (unsigned)(uintptr_t)&Bs[1][0]};

  v8f acc[MT][NT];
#pragma unroll
  for (int i = 0; i < MT; ++i)
#pragma unroll
    for (int j = 0; j < NT; ++j)
#pragma unroll
      for (int r = 0; r < 8; ++r) acc[i][j][r] = 0.0f;

  // Prologue: DMA first tile pair into buffer 0 (wave 0 issues; EXEC-independent)
  if (wave == 0) {
    tdm_load_tile_2d(ldsA[0], Vb, BM, BK, NPIX);
    tdm_load_tile_2d(ldsB[0], Kb, BN, BK, NPIX);
  }

  int buf = 0;
  for (int kk = 0; kk < NPIX; kk += BK) {
    if (wave == 0) {
      if (kk + BK < NPIX) {
        // DMA next tile pair into the other buffer, then wait until the
        // current pair (issued earlier; TDM is in-order) has landed.
        tdm_load_tile_2d(ldsA[buf ^ 1], Vb + kk + BK, BM, BK, NPIX);
        tdm_load_tile_2d(ldsB[buf ^ 1], Kb + kk + BK, BN, BK, NPIX);
        __builtin_amdgcn_s_wait_tensorcnt(2);
      } else {
        __builtin_amdgcn_s_wait_tensorcnt(0);
      }
    }
    __syncthreads();
    mma_step<MT, NT>(As[buf], Bs[buf], m0, n0, acc);
    __syncthreads();   // all waves done reading before this buffer is re-DMA'd
    buf ^= 1;
  }

  const int lane = tid & 31, g = lane >> 4, lr = lane & 15;
  __bf16* Ob = KVt + ((size_t)br * NBATCH + bz) * CCH * MCH;
#pragma unroll
  for (int i = 0; i < MT; ++i)
#pragma unroll
    for (int j = 0; j < NT; ++j)
#pragma unroll
      for (int r = 0; r < 8; ++r) {
        int row = row0 + m0 + i * 16 + g * 8 + r;
        int col = n0 + j * 16 + lr;
        Ob[(size_t)row * MCH + col] = f2bf(acc[i][j][r]);
      }
}

// ---------------------------------------------------------------------------
// wv[b] (C x N) = sum_br KVt[br,b] (C x 64) @ (Q[b] * normv[br,b,:]) (64 x N)
// ---------------------------------------------------------------------------
__global__ __launch_bounds__(256) void gemm_branch_kernel(
    const __bf16* __restrict__ KVt, const __bf16* __restrict__ Q,
    const float* __restrict__ normv, __bf16* __restrict__ wv) {
  constexpr int BM = 128, BN = 128, BK = 32, MT = 2, NT = 4, WAVES_N = 2;
  __shared__ alignas(16) __bf16 As[BM * LDK];
  __shared__ alignas(16) __bf16 Bs[BN * LDK];

  const int col0 = blockIdx.x * BN;
  const int row0 = blockIdx.y * BM;
  const int bz   = blockIdx.z;
  const int tid  = threadIdx.x;
  const int wave = tid >> 5;
  const int m0   = (wave / WAVES_N) * 32;
  const int n0   = (wave % WAVES_N) * (16 * NT);
  const __bf16* Qb = Q + (size_t)bz * MCH * NPIX;

  v8f acc[MT][NT];
#pragma unroll
  for (int i = 0; i < MT; ++i)
#pragma unroll
    for (int j = 0; j < NT; ++j)
#pragma unroll
      for (int r = 0; r < 8; ++r) acc[i][j][r] = 0.0f;

  for (int br = 0; br < 3; ++br) {
    const __bf16* Ab = KVt + ((size_t)br * NBATCH + bz) * CCH * MCH;
    const float*  nv = normv + ((size_t)br * NBATCH + bz) * NPIX;
    for (int kk = 0; kk < MCH; kk += BK) {
      // A tile: 16-byte bf16 copies
      for (int idx = tid; idx < BM * (BK / 8); idx += 256) {
        int r = idx >> 2, k8 = (idx & 3) << 3;
        *(v8bf*)(As + r * LDK + k8) =
            *(const v8bf*)(Ab + (size_t)(row0 + r) * MCH + kk + k8);
      }
      // B tile: vector Q/norm loads over n, norm-scaled, scalar LDS stores
      for (int idx = tid; idx < BK * (BN / 8); idx += 256) {
        int k = idx >> 4, n8 = (idx & 15) << 3;
        v8bf qv = *(const v8bf*)(Qb + (size_t)(kk + k) * NPIX + col0 + n8);
        float4 nv0 = *(const float4*)(nv + col0 + n8);
        float4 nv1 = *(const float4*)(nv + col0 + n8 + 4);
        float nn[8] = {nv0.x, nv0.y, nv0.z, nv0.w, nv1.x, nv1.y, nv1.z, nv1.w};
#pragma unroll
        for (int i = 0; i < 8; ++i)
          Bs[(n8 + i) * LDK + k] = f2bf(bf2f(qv[i]) * nn[i]);
      }
      __syncthreads();
      mma_step<MT, NT>(As, Bs, m0, n0, acc);
      __syncthreads();
    }
  }

  const int lane = tid & 31, g = lane >> 4, lr = lane & 15;
  __bf16* Ob = wv + (size_t)bz * CCH * NPIX;
#pragma unroll
  for (int i = 0; i < MT; ++i)
#pragma unroll
    for (int j = 0; j < NT; ++j)
#pragma unroll
      for (int r = 0; r < 8; ++r) {
        int row = row0 + m0 + i * 16 + g * 8 + r;
        int col = col0 + n0 + j * 16 + lr;
        Ob[(size_t)row * NPIX + col] = f2bf(acc[i][j][r]);
      }
}

// ---------------------------------------------------------------------------
// out[b] = x[b] + gamma * (Wl (C x C) @ wv[b] (C x N) + bl)
// ---------------------------------------------------------------------------
__global__ __launch_bounds__(256) void gemm_final_kernel(
    const float* __restrict__ Wl, const __bf16* __restrict__ wv,
    const float* __restrict__ bl, const float* __restrict__ x,
    const float* __restrict__ gamma, float* __restrict__ out) {
  constexpr int BM = 128, BN = 128, BK = 32, MT = 2, NT = 4, WAVES_N = 2;
  __shared__ alignas(16) __bf16 As[BM * LDK];
  __shared__ alignas(16) __bf16 Bs[BN * LDK];

  const int col0 = blockIdx.x * BN;
  const int row0 = blockIdx.y * BM;
  const int bz   = blockIdx.z;
  const int tid  = threadIdx.x;
  const int wave = tid >> 5;
  const int m0   = (wave / WAVES_N) * 32;
  const int n0   = (wave % WAVES_N) * (16 * NT);
  const __bf16* Wb = wv + (size_t)bz * CCH * NPIX;

  v8f acc[MT][NT];
#pragma unroll
  for (int i = 0; i < MT; ++i)
#pragma unroll
    for (int j = 0; j < NT; ++j)
#pragma unroll
      for (int r = 0; r < 8; ++r) acc[i][j][r] = 0.0f;

  for (int kk = 0; kk < CCH; kk += BK) {
    // A tile: Wl fp32 -> bf16, float4 loads
    for (int idx = tid; idx < BM * (BK / 4); idx += 256) {
      int r = idx >> 3, k4 = (idx & 7) << 2;
      float4 w4 = *(const float4*)(Wl + (size_t)(row0 + r) * CCH + kk + k4);
      v4bf o; o[0] = f2bf(w4.x); o[1] = f2bf(w4.y); o[2] = f2bf(w4.z); o[3] = f2bf(w4.w);
      *(v4bf*)(As + r * LDK + k4) = o;
    }
    // B tile: wv bf16, 16-byte vector loads over n, scalar LDS stores
    for (int idx = tid; idx < BK * (BN / 8); idx += 256) {
      int k = idx >> 4, n8 = (idx & 15) << 3;
      v8bf w8 = *(const v8bf*)(Wb + (size_t)(kk + k) * NPIX + col0 + n8);
#pragma unroll
      for (int i = 0; i < 8; ++i) Bs[(n8 + i) * LDK + k] = w8[i];
    }
    if (kk + BK < CCH) {
      __builtin_prefetch(Wl + (size_t)(row0 + (tid >> 3)) * CCH + kk + BK + ((tid & 7) << 2), 0, 1);
      __builtin_prefetch(Wb + (size_t)(kk + BK + (tid >> 4)) * NPIX + col0 + ((tid & 15) << 3), 0, 1);
    }
    __syncthreads();
    mma_step<MT, NT>(As, Bs, m0, n0, acc);
    __syncthreads();
  }

  const int lane = tid & 31, g = lane >> 4, lr = lane & 15;
  const float gm = gamma[0];
#pragma unroll
  for (int i = 0; i < MT; ++i)
#pragma unroll
    for (int j = 0; j < NT; ++j)
#pragma unroll
      for (int r = 0; r < 8; ++r) {
        int row = row0 + m0 + i * 16 + g * 8 + r;
        int col = col0 + n0 + j * 16 + lr;
        size_t o = (size_t)bz * CCH * NPIX + (size_t)row * NPIX + col;
        out[o] = x[o] + gm * (acc[i][j][r] + bl[row]);
      }
}

// ---------------------------------------------------------------------------
extern "C" void kernel_launch(void* const* d_in, const int* in_sizes, int n_in,
                              void* d_out, int out_size, void* d_ws, size_t ws_size,
                              hipStream_t stream) {
  (void)in_sizes; (void)n_in; (void)out_size; (void)ws_size;
  const float* x     = (const float*)d_in[0];
  const float* y     = (const float*)d_in[1];
  const float* z     = (const float*)d_in[2];
  const float* Wq    = (const float*)d_in[3];
  const float* bq    = (const float*)d_in[4];
  const float* Wk1   = (const float*)d_in[5];
  const float* bk1   = (const float*)d_in[6];
  const float* Wk2   = (const float*)d_in[7];
  const float* bk2   = (const float*)d_in[8];
  const float* Wk3   = (const float*)d_in[9];
  const float* bk3   = (const float*)d_in[10];
  const float* Wv    = (const float*)d_in[11];
  const float* bv    = (const float*)d_in[12];
  const float* Wl    = (const float*)d_in[13];
  const float* bl    = (const float*)d_in[14];
  const float* gamma = (const float*)d_in[15];

  auto au = [](size_t v) { return (v + 255) & ~(size_t)255; };
  const size_t elQ = (size_t)NBATCH * MCH * NPIX;   // Q-sized tensors (bf16)
  const size_t elV = (size_t)NBATCH * CCH * NPIX;   // V-sized tensors (bf16)
  char* w = (char*)d_ws;
  __bf16* Qb   = (__bf16*)w;  w += au(elQ * sizeof(__bf16));
  __bf16* Kall = (__bf16*)w;  w += au(3 * elQ * sizeof(__bf16));
  __bf16* Vb   = (__bf16*)w;  w += au(elV * sizeof(__bf16));
  __bf16* KVt  = (__bf16*)w;  w += au((size_t)3 * NBATCH * CCH * MCH * sizeof(__bf16));
  float*  Ksum = (float*)w;   w += au((size_t)3 * NBATCH * MCH * sizeof(float));
  float*  nrm  = (float*)w;   w += au((size_t)3 * NBATCH * NPIX * sizeof(float));
  __bf16* wvb  = (__bf16*)w;  w += au(elV * sizeof(__bf16));

  const dim3 blk(256);
  // Projections (bias + elus feature map fused; bf16 outputs)
  gemm_wx_kernel<64, 128, true ><<<dim3(32, 1, NBATCH), blk, 0, stream>>>(Wq,  x, bq,  Qb,            MCH, CCH);
  gemm_wx_kernel<64, 128, true ><<<dim3(32, 1, NBATCH), blk, 0, stream>>>(Wk1, x, bk1, Kall,          MCH, CCH);
  gemm_wx_kernel<64, 128, true ><<<dim3(32, 1, NBATCH), blk, 0, stream>>>(Wk2, y, bk2, Kall + elQ,    MCH, CCH);
  gemm_wx_kernel<64, 128, true ><<<dim3(32, 1, NBATCH), blk, 0, stream>>>(Wk3, z, bk3, Kall + 2*elQ,  MCH, CCH);
  gemm_wx_kernel<128, 128, false><<<dim3(32, 4, NBATCH), blk, 0, stream>>>(Wv, x, bv,  Vb,            CCH, CCH);
  // K row sums (+EPS), per-pixel normalizers
  ksum_kernel<<<3 * NBATCH * MCH, 128, 0, stream>>>(Kall, Ksum);
  norm_kernel<<<(3 * NBATCH * NPIX) / 256, 256, 0, stream>>>(Qb, Ksum, nrm);
  // KV^T per branch (TDM-staged, double buffered), then rank-64 expansion
  gemm_kvt_kernel<<<dim3(4, 3, NBATCH), blk, 0, stream>>>(Vb, Kall, KVt);
  gemm_branch_kernel<<<dim3(32, 4, NBATCH), blk, 0, stream>>>(KVt, Qb, nrm, wvb);
  // Final 1x1 conv + residual: out = x + gamma*(Wl@wv + bl)
  gemm_final_kernel<<<dim3(32, 4, NBATCH), blk, 0, stream>>>(Wl, wvb, bl, x, gamma, (float*)d_out);
}
